// VisionTransformer_12446815224190
// MI455X (gfx1250) — matrix operand order repaired and checked by reading, not compiled, not run on hardware
//
#include <hip/hip_runtime.h>
#include <hip/hip_bf16.h>

// ---------------------------------------------------------------------------
// Complex ViT forward for MI455X (gfx1250, wave32, WMMA).
// All complex GEMMs = 4 real bf16 WMMA GEMMs (f32 accumulate), weights
// pre-converted to de-interleaved bf16 planes in workspace. GEMM uses a
// 64x128 block tile, 32x32 per wave (16 WMMA / 32-k substep), K-slab 64
// with ping-pong LDS buffers and software-pipelined global->LDS staging.
// ---------------------------------------------------------------------------

#define kB   8
#define kC   3
#define kHW  224
#define kP   16
#define kE   768
#define kNH  12
#define kHID 3072
#define kL   8
#define kNC  1000
#define kHD  64          // kE / kNH
#define kG   14          // kHW / kP
#define kNP  196         // kG*kG
#define kS   197         // kNP + 1
#define kMR  (kB * kS)   // 1576
#define kPR  (kB * kNP)  // 1568
#define kEPS 1e-5f

#define BK   64          // K-slab per LDS stage
#define AS   72          // LDS row stride in bf16 (64 + 8 pad, 144B rows)

typedef __attribute__((ext_vector_type(16))) __bf16 v16bf;
typedef __attribute__((ext_vector_type(8)))  float  v8f;

union Frag { v16bf v; unsigned u[8]; };
union Acc  { v8f  v; float    f[8]; };

__device__ __forceinline__ unsigned f2bf(float x) {
  unsigned u = __float_as_uint(x);
  return (u + 0x7FFFu + ((u >> 16) & 1u)) >> 16;   // round-to-nearest-even
}
__device__ __forceinline__ unsigned pack2bf(float lo, float hi) {
  return f2bf(lo) | (f2bf(hi) << 16);
}
__device__ __forceinline__ float gelu_tanh(float x) {
  float x3 = x * x * x;
  return 0.5f * x * (1.0f + tanhf(0.7978845608028654f * (x + 0.044715f * x3)));
}
__device__ __forceinline__ v8f wmma_bf16(const Frag& a, const Frag& b, v8f c) {
  return __builtin_amdgcn_wmma_f32_16x16x32_bf16(false, a.v, false, b.v,
                                                 (short)0, c, false, false);
}

// ---------------------------------------------------------------------------
// Weight conversion: interleaved complex f32 -> bf16 re/im planes
// ---------------------------------------------------------------------------
__global__ void cvt_cplx_bf16_kernel(const float* __restrict__ src,
                                     unsigned short* __restrict__ dre,
                                     unsigned short* __restrict__ dim,
                                     long long n) {
  long long i = (long long)blockIdx.x * blockDim.x + threadIdx.x;
  long long stride = (long long)gridDim.x * blockDim.x;
  for (; i < n; i += stride) {
    dre[i] = (unsigned short)f2bf(src[2 * i]);
    dim[i] = (unsigned short)f2bf(src[2 * i + 1]);
  }
}

// ---------------------------------------------------------------------------
// Generic complex GEMM: C[M,N] = A[M,K] @ W^T + bias (+gelu) (+residual)
//   A: interleaved complex f32, row-major, lda complex elems
//   Wre/Wim: bf16 planes, row-major [N, K]
//   Block: 256 thr = 8 waves; block tile 64(M) x 128(N); wave tile 32x32.
//   N must be a multiple of 128 (true for 768/2304/3072).
// ---------------------------------------------------------------------------
template <bool GELU, bool RESID>
__global__ __launch_bounds__(256)
void cgemm_kernel(const float* __restrict__ A, int lda,
                  const unsigned short* __restrict__ Wre,
                  const unsigned short* __restrict__ Wim, int ldw,
                  const float* __restrict__ bias,
                  float* __restrict__ Cout, int ldc,
                  const float* __restrict__ Cres,
                  int M, int K) {
  __shared__ unsigned short AreS[2][64 * AS];   // ping-pong bf16 planes
  __shared__ unsigned short AimS[2][64 * AS];

  const int tid  = threadIdx.x;
  const int lane = tid & 31;
  const int wv   = tid >> 5;
  const int mw   = wv & 1;    // 0..1 -> 32-row half of block tile
  const int nw   = wv >> 1;   // 0..3 -> 32-col quarter of block tile
  const int m0   = blockIdx.y * 64;
  const int n0   = blockIdx.x * 128;
  const int l15  = lane & 15;
  const int hi   = (lane < 16) ? 0 : 1;

  // ---- staging helpers: 64 rows x 64 k complex slab, 8 float4 / thread ----
  auto loadSlab = [&](int kb, float4* nx) {
#pragma unroll
    for (int i = 0; i < 8; ++i) {
      int e   = tid + i * 256;
      int row = e >> 5;
      int kp  = (e & 31) << 1;         // even k within slab
      int gr  = m0 + row;
      float4 v = make_float4(0.f, 0.f, 0.f, 0.f);
      if (gr < M) v = *(const float4*)(A + ((size_t)gr * lda + kb + kp) * 2);
      nx[i] = v;
    }
  };
  auto storeSlab = [&](int pb, const float4* nx) {
#pragma unroll
    for (int i = 0; i < 8; ++i) {
      int e   = tid + i * 256;
      int row = e >> 5;
      int kp  = (e & 31) << 1;
      *(unsigned*)&AreS[pb][row * AS + kp] = pack2bf(nx[i].x, nx[i].z);
      *(unsigned*)&AimS[pb][row * AS + kp] = pack2bf(nx[i].y, nx[i].w);
    }
  };

  Acc cre[2][2], cim[2][2];
#pragma unroll
  for (int r = 0; r < 2; ++r)
#pragma unroll
    for (int c = 0; c < 2; ++c)
#pragma unroll
      for (int j = 0; j < 8; ++j) { cre[r][c].f[j] = 0.f; cim[r][c].f[j] = 0.f; }

  // ---- compute one 64-k slab from LDS buffer pb (2 substeps of 32) ----
  auto computeSlab = [&](int pb, int kb) {
#pragma unroll
    for (int ks = 0; ks < BK; ks += 32) {
      // A fragments: 16x32 bf16 layout (lanes<16: K{0..15}; else K{16..31})
      Frag are[2], aim[2], aimn[2];
#pragma unroll
      for (int r = 0; r < 2; ++r) {
        int row = mw * 32 + r * 16 + l15;
        const unsigned short* br_ = &AreS[pb][row * AS + ks + hi * 16];
        const unsigned short* bi_ = &AimS[pb][row * AS + ks + hi * 16];
        const unsigned* pr0 = (const unsigned*)br_;
        const unsigned* pr1 = (const unsigned*)(br_ + 8);
        const unsigned* pi0 = (const unsigned*)bi_;
        const unsigned* pi1 = (const unsigned*)(bi_ + 8);
#pragma unroll
        for (int j = 0; j < 4; ++j) {
          are[r].u[j] = pr0[j]; are[r].u[4 + j] = pr1[j];
          aim[r].u[j] = pi0[j]; aim[r].u[4 + j] = pi1[j];
        }
#pragma unroll
        for (int j = 0; j < 8; ++j) aimn[r].u[j] = aim[r].u[j] ^ 0x80008000u;
      }
      // B fragments from global bf16 planes (32x16 layout), reused across r
#pragma unroll
      for (int c = 0; c < 2; ++c) {
        int col = n0 + nw * 32 + c * 16 + l15;
        int kk  = kb + ks + hi * 16;
        const uint4* qr = (const uint4*)(Wre + (size_t)col * ldw + kk);
        const uint4* qi = (const uint4*)(Wim + (size_t)col * ldw + kk);
        uint4 r0 = qr[0], r1 = qr[1], i0 = qi[0], i1 = qi[1];
        Frag bre, bim;
        bre.u[0] = r0.x; bre.u[1] = r0.y; bre.u[2] = r0.z; bre.u[3] = r0.w;
        bre.u[4] = r1.x; bre.u[5] = r1.y; bre.u[6] = r1.z; bre.u[7] = r1.w;
        bim.u[0] = i0.x; bim.u[1] = i0.y; bim.u[2] = i0.z; bim.u[3] = i0.w;
        bim.u[4] = i1.x; bim.u[5] = i1.y; bim.u[6] = i1.z; bim.u[7] = i1.w;
#pragma unroll
        for (int r = 0; r < 2; ++r) {
          cre[r][c].v = wmma_bf16(are[r],  bre, cre[r][c].v);  // + Are*Wre
          cre[r][c].v = wmma_bf16(aimn[r], bim, cre[r][c].v);  // - Aim*Wim
          cim[r][c].v = wmma_bf16(are[r],  bim, cim[r][c].v);  // + Are*Wim
          cim[r][c].v = wmma_bf16(aim[r],  bre, cim[r][c].v);  // + Aim*Wre
        }
      }
    }
  };

  // ---- software-pipelined main loop (ping-pong LDS) ----
  {
    float4 cur[8];
    loadSlab(0, cur);
    storeSlab(0, cur);
  }
  __syncthreads();
  int pb = 0;
  for (int kb = 0; kb < K; kb += BK) {
    float4 nx[8];
    const bool more = (kb + BK) < K;
    if (more) loadSlab(kb + BK, nx);     // issue loads, latency hides in WMMAs
    computeSlab(pb, kb);
    if (more) storeSlab(pb ^ 1, nx);
    __syncthreads();
    pb ^= 1;
  }

  // ---- epilogue: bias (+gelu) (+residual), interleaved complex f32 out ----
#pragma unroll
  for (int r = 0; r < 2; ++r) {
#pragma unroll
    for (int c = 0; c < 2; ++c) {
      int col = n0 + nw * 32 + c * 16 + l15;
      float br = 0.f, bi = 0.f;
      if (bias) { br = bias[2 * col]; bi = bias[2 * col + 1]; }
#pragma unroll
      for (int j = 0; j < 8; ++j) {
        int row = m0 + mw * 32 + r * 16 + j + hi * 8;
        if (row < M) {
          float re = cre[r][c].f[j] + br;
          float im = cim[r][c].f[j] + bi;
          if (GELU) { re = gelu_tanh(re); im = gelu_tanh(im); }
          size_t o = ((size_t)row * ldc + col) * 2;
          if (RESID) { re += Cres[o]; im += Cres[o + 1]; }
          Cout[o]     = re;
          Cout[o + 1] = im;
        }
      }
    }
  }
}

// ---------------------------------------------------------------------------
// im2row patch gather: (B,C,224,224) re/im planes -> [1568, 768] complex
// ---------------------------------------------------------------------------
__global__ void im2row_kernel(const float* __restrict__ xre,
                              const float* __restrict__ xim,
                              float* __restrict__ patches) {
  long long idx = (long long)blockIdx.x * blockDim.x + threadIdx.x;
  const long long total = (long long)kPR * kE;
  if (idx >= total) return;
  int r = (int)(idx / kE), j = (int)(idx % kE);
  int b = r / kNP, pp = r % kNP;
  int gy = pp / kG, gx = pp % kG;
  int c = j / (kP * kP), rem = j % (kP * kP);
  int py = rem / kP, px = rem % kP;
  size_t s = (((size_t)(b * kC + c) * kHW) + gy * kP + py) * kHW + gx * kP + px;
  patches[2 * idx]     = xre[s];
  patches[2 * idx + 1] = xim[s];
}

// ---------------------------------------------------------------------------
// Token assembly: x[b,0]=cls+pos[0]; x[b,1+n]=embtmp+in_b+pos[1+n]
// ---------------------------------------------------------------------------
__global__ void assemble_kernel(const float* __restrict__ embtmp,
                                const float* __restrict__ inb,
                                const float* __restrict__ cls,
                                const float* __restrict__ pos,
                                float* __restrict__ x) {
  long long idx = (long long)blockIdx.x * blockDim.x + threadIdx.x;
  const long long total = (long long)kMR * kE;
  if (idx >= total) return;
  int row = (int)(idx / kE), j = (int)(idx % kE);
  int b = row / kS, s = row % kS;
  size_t po = ((size_t)s * kE + j) * 2;
  float pr = pos[po], pi = pos[po + 1];
  float vr, vi;
  if (s == 0) {
    vr = cls[2 * j] + pr;
    vi = cls[2 * j + 1] + pi;
  } else {
    size_t e = ((size_t)(b * kNP + (s - 1)) * kE + j) * 2;
    vr = embtmp[e]     + inb[2 * j]     + pr;
    vi = embtmp[e + 1] + inb[2 * j + 1] + pi;
  }
  x[2 * idx]     = vr;
  x[2 * idx + 1] = vi;
  (void)b;
}

// ---------------------------------------------------------------------------
// rmsnorm over E complex elems, times complex weight w: one block per row
// ---------------------------------------------------------------------------
__global__ __launch_bounds__(256)
void rmsnorm_kernel(const float* __restrict__ X, long long rowStrideC,
                    const float* __restrict__ w,
                    float* __restrict__ Y, long long yStrideC) {
  __shared__ float red[256];
  const int row = blockIdx.x, tid = threadIdx.x;
  const float* xp = X + (size_t)row * rowStrideC * 2;
  float s = 0.f;
  for (int j = tid; j < kE; j += 256) {
    float a = xp[2 * j], b = xp[2 * j + 1];
    s += a * a + b * b;
  }
  red[tid] = s;
  __syncthreads();
  for (int st = 128; st > 0; st >>= 1) {
    if (tid < st) red[tid] += red[tid + st];
    __syncthreads();
  }
  float inv = rsqrtf(red[0] * (1.0f / kE) + kEPS);
  float* yp = Y + (size_t)row * yStrideC * 2;
  for (int j = tid; j < kE; j += 256) {
    float a = xp[2 * j] * inv, b = xp[2 * j + 1] * inv;
    float wr = w[2 * j], wi = w[2 * j + 1];
    yp[2 * j]     = a * wr - b * wi;
    yp[2 * j + 1] = a * wi + b * wr;
  }
}

// ---------------------------------------------------------------------------
// Attention: one workgroup per (b,h). q/k rmsnorm (complex qn/kn weights),
// scores = Re(q conj(k)) / 8, softmax, o = attn @ v, x += o.
// q/k/v staged as f32 planes in ~304KB dynamic LDS (320KB WGP LDS).
// ---------------------------------------------------------------------------
__global__ __launch_bounds__(256)
void attn_kernel(const float* __restrict__ qkvb, float* __restrict__ x,
                 const float* __restrict__ qn, const float* __restrict__ kn) {
  extern __shared__ float sm[];
  float* qre  = sm;
  float* qim  = qre + kS * kHD;
  float* kre  = qim + kS * kHD;
  float* kim  = kre + kS * kHD;
  float* vre  = kim + kS * kHD;
  float* vim  = vre + kS * kHD;
  float* invq = vim + kS * kHD;
  float* invk = invq + kS;

  const int b = blockIdx.x / kNH;
  const int h = blockIdx.x % kNH;
  const int tid = threadIdx.x;
  const size_t rowF = (size_t)3 * kE * 2;  // floats per token row of qkv buf
  const float* basep = qkvb + (size_t)b * kS * rowF + (size_t)h * kHD * 2;

  // pass 1: per-row mean|.|^2 for q and k
  for (int n = tid; n < kS; n += blockDim.x) {
    const float* qp = basep + (size_t)n * rowF;
    const float* kp = qp + kE * 2;
    float sq = 0.f, sk = 0.f;
    for (int d = 0; d < kHD; ++d) {
      float a = qp[2 * d], c = qp[2 * d + 1]; sq += a * a + c * c;
      float e = kp[2 * d], f = kp[2 * d + 1]; sk += e * e + f * f;
    }
    invq[n] = rsqrtf(sq * (1.0f / kHD) + kEPS);
    invk[n] = rsqrtf(sk * (1.0f / kHD) + kEPS);
  }
  __syncthreads();

  // pass 2: normalized q,k (complex-weighted) and v into LDS planes
  for (int i = tid; i < kS * kHD; i += blockDim.x) {
    int n = i / kHD, d = i % kHD;
    const float* qp = basep + (size_t)n * rowF + 2 * d;
    float qr = qp[0], qi = qp[1];
    float wr = qn[2 * d], wi = qn[2 * d + 1];
    float s = invq[n];
    qre[i] = (qr * wr - qi * wi) * s;
    qim[i] = (qr * wi + qi * wr) * s;
    float kr = qp[2 * kE], ki = qp[2 * kE + 1];
    wr = kn[2 * d]; wi = kn[2 * d + 1]; s = invk[n];
    kre[i] = (kr * wr - ki * wi) * s;
    kim[i] = (kr * wi + ki * wr) * s;
    vre[i] = qp[4 * kE];
    vim[i] = qp[4 * kE + 1];
  }
  __syncthreads();

  // pass 3: one thread per query row
  const int t = tid;
  if (t < kS) {
    const float scale = 0.125f;   // HD^-0.5
    const float* qrp = qre + t * kHD;
    const float* qip = qim + t * kHD;
    float mx = -3.4e38f;
    for (int m = 0; m < kS; ++m) {
      const float* krp = kre + m * kHD;
      const float* kip = kim + m * kHD;
      float s = 0.f;
#pragma unroll 16
      for (int d = 0; d < kHD; ++d) s += qrp[d] * krp[d] + qip[d] * kip[d];
      mx = fmaxf(mx, s * scale);
    }
    float ore[kHD], oim[kHD];
#pragma unroll
    for (int d = 0; d < kHD; ++d) { ore[d] = 0.f; oim[d] = 0.f; }
    float sum = 0.f;
    for (int m = 0; m < kS; ++m) {
      const float* krp = kre + m * kHD;
      const float* kip = kim + m * kHD;
      float s = 0.f;
#pragma unroll 16
      for (int d = 0; d < kHD; ++d) s += qrp[d] * krp[d] + qip[d] * kip[d];
      float p = __expf(s * scale - mx);
      sum += p;
      const float* vrp = vre + m * kHD;
      const float* vip = vim + m * kHD;
#pragma unroll
      for (int d = 0; d < kHD; ++d) { ore[d] += p * vrp[d]; oim[d] += p * vip[d]; }
    }
    float inv = 1.0f / sum;
    float* xp = x + ((size_t)(b * kS + t) * kE + h * kHD) * 2;
#pragma unroll
    for (int d = 0; d < kHD; ++d) {
      xp[2 * d]     += ore[d] * inv;
      xp[2 * d + 1] += oim[d] * inv;
    }
  }
}

// ---------------------------------------------------------------------------
// Head: out[8,1000] = clsn @ head_w^T + head_b (tiny; plain VALU, f32)
// ---------------------------------------------------------------------------
__global__ void head_gemm_kernel(const float* __restrict__ clsn,
                                 const float* __restrict__ W,
                                 const float* __restrict__ bias,
                                 float* __restrict__ out) {
  int idx = blockIdx.x * blockDim.x + threadIdx.x;
  if (idx >= kB * kNC) return;
  int m = idx / kNC, n = idx % kNC;
  const float* a = clsn + (size_t)m * kE * 2;
  const float* w = W + (size_t)n * kE * 2;
  float cr = bias[2 * n], ci = bias[2 * n + 1];
  for (int k = 0; k < kE; ++k) {
    float ar = a[2 * k], ai = a[2 * k + 1];
    float br = w[2 * k], bi = w[2 * k + 1];
    cr += ar * br - ai * bi;
    ci += ar * bi + ai * br;
  }
  out[2 * idx]     = cr;
  out[2 * idx + 1] = ci;
}

// ---------------------------------------------------------------------------
extern "C" void kernel_launch(void* const* d_in, const int* in_sizes, int n_in,
                              void* d_out, int out_size, void* d_ws, size_t ws_size,
                              hipStream_t stream) {
  (void)in_sizes; (void)out_size; (void)ws_size;
  if (n_in < 18) return;

  const float* x_re      = (const float*)d_in[0];
  const float* x_im      = (const float*)d_in[1];
  const float* in_w      = (const float*)d_in[2];
  const float* in_b      = (const float*)d_in[3];
  const float* cls_token = (const float*)d_in[4];
  const float* pos_emb   = (const float*)d_in[5];
  const float* qkv_w     = (const float*)d_in[6];
  const float* qkv_b     = (const float*)d_in[7];
  const float* qn_w      = (const float*)d_in[8];
  const float* kn_w      = (const float*)d_in[9];
  const float* ln_w      = (const float*)d_in[10];
  const float* fc1_w     = (const float*)d_in[11];
  const float* fc1_b     = (const float*)d_in[12];
  const float* fc2_w     = (const float*)d_in[13];
  const float* fc2_b     = (const float*)d_in[14];
  const float* head_ln_w = (const float*)d_in[15];
  const float* head_w    = (const float*)d_in[16];
  const float* head_b    = (const float*)d_in[17];

  // ---- workspace carve (256B aligned) ----
  char* ws = (char*)d_ws;
  size_t off = 0;
  auto take = [&](size_t bytes) -> void* {
    void* p = ws + off;
    off += (bytes + 255) & ~(size_t)255;
    return p;
  };
  unsigned short* inw_re  = (unsigned short*)take((size_t)kE * kE * 2);
  unsigned short* inw_im  = (unsigned short*)take((size_t)kE * kE * 2);
  unsigned short* qkvw_re = (unsigned short*)take((size_t)kL * 3 * kE * kE * 2);
  unsigned short* qkvw_im = (unsigned short*)take((size_t)kL * 3 * kE * kE * 2);
  unsigned short* fc1w_re = (unsigned short*)take((size_t)kL * kHID * kE * 2);
  unsigned short* fc1w_im = (unsigned short*)take((size_t)kL * kHID * kE * 2);
  unsigned short* fc2w_re = (unsigned short*)take((size_t)kL * kE * kHID * 2);
  unsigned short* fc2w_im = (unsigned short*)take((size_t)kL * kE * kHID * 2);
  float* patches = (float*)take((size_t)kPR * kE * 2 * 4);
  float* embtmp  = (float*)take((size_t)kPR * kE * 2 * 4);
  float* xbuf    = (float*)take((size_t)kMR * kE * 2 * 4);
  float* qkvb    = (float*)take((size_t)kMR * 3 * kE * 2 * 4);
  float* ybuf    = (float*)take((size_t)kMR * kE * 2 * 4);
  float* hbuf    = (float*)take((size_t)kMR * kHID * 2 * 4);
  float* clsn    = (float*)take((size_t)kB * kE * 2 * 4);

  // ---- weights -> bf16 planes ----
  cvt_cplx_bf16_kernel<<<1024, 256, 0, stream>>>(in_w,  inw_re,  inw_im,  (long long)kE * kE);
  cvt_cplx_bf16_kernel<<<4096, 256, 0, stream>>>(qkv_w, qkvw_re, qkvw_im, (long long)kL * 3 * kE * kE);
  cvt_cplx_bf16_kernel<<<4096, 256, 0, stream>>>(fc1_w, fc1w_re, fc1w_im, (long long)kL * kHID * kE);
  cvt_cplx_bf16_kernel<<<4096, 256, 0, stream>>>(fc2_w, fc2w_re, fc2w_im, (long long)kL * kE * kHID);

  // ---- patch embedding ----
  im2row_kernel<<<((long long)kPR * kE + 255) / 256, 256, 0, stream>>>(x_re, x_im, patches);
  cgemm_kernel<false, false><<<dim3(kE / 128, (kPR + 63) / 64), 256, 0, stream>>>(
      patches, kE, inw_re, inw_im, kE, nullptr, embtmp, kE, nullptr, kPR, kE);
  assemble_kernel<<<((long long)kMR * kE + 255) / 256, 256, 0, stream>>>(
      embtmp, in_b, cls_token, pos_emb, xbuf);

  // ---- transformer layers ----
  const int attn_smem = (6 * kS * kHD + 2 * kS) * (int)sizeof(float);  // ~304 KB
  for (int l = 0; l < kL; ++l) {
    cgemm_kernel<false, false><<<dim3(3 * kE / 128, (kMR + 63) / 64), 256, 0, stream>>>(
        xbuf, kE,
        qkvw_re + (size_t)l * 3 * kE * kE, qkvw_im + (size_t)l * 3 * kE * kE, kE,
        qkv_b + (size_t)l * 3 * kE * 2,
        qkvb, 3 * kE, nullptr, kMR, kE);

    attn_kernel<<<kB * kNH, 256, attn_smem, stream>>>(
        qkvb, xbuf, qn_w + (size_t)l * kHD * 2, kn_w + (size_t)l * kHD * 2);

    rmsnorm_kernel<<<kMR, 256, 0, stream>>>(xbuf, kE, ln_w + (size_t)l * kE * 2, ybuf, kE);

    cgemm_kernel<true, false><<<dim3(kHID / 128, (kMR + 63) / 64), 256, 0, stream>>>(
        ybuf, kE,
        fc1w_re + (size_t)l * kHID * kE, fc1w_im + (size_t)l * kHID * kE, kE,
        fc1_b + (size_t)l * kHID * 2,
        hbuf, kHID, nullptr, kMR, kE);

    cgemm_kernel<false, true><<<dim3(kE / 128, (kMR + 63) / 64), 256, 0, stream>>>(
        hbuf, kHID,
        fc2w_re + (size_t)l * kE * kHID, fc2w_im + (size_t)l * kE * kHID, kHID,
        fc2_b + (size_t)l * kE * 2,
        xbuf, kE, xbuf, kMR, kHID);
  }

  // ---- head ----
  rmsnorm_kernel<<<kB, 256, 0, stream>>>(xbuf, (long long)kS * kE, head_ln_w, clsn, kE);
  head_gemm_kernel<<<(kB * kNC + 255) / 256, 256, 0, stream>>>(
      clsn, head_w, head_b, (float*)d_out);
}
